// CrossNonLocal2D_6648609375054
// MI455X (gfx1250) — matrix operand order repaired and checked by reading, not compiled, not run on hardware
//
#include <hip/hip_runtime.h>
#include <hip/hip_bf16.h>

// CrossNonLocal2D for MI455X (gfx1250): f16 WMMA (16x16x32) with f32 accum,
// flash-attention style online softmax to avoid materializing the 536MB attn.
//
// Shapes: B=8, C=256, I=128, H=W=64 -> N=4096.
// ws layout (bytes):
//   [0, 256KB)  : f16 weights (theta_w, phi_w, g_w, out_w), 4 x 32768 f16
//   thetaT [B,N,I] f16 (8MB), phiT [B,N,I] f16 (8MB), g [B,I,N] f16 (8MB),
//   yT [B,N,I] f16 (8MB).  Total ~34MB (assumes ws_size is sufficient).

typedef _Float16 f16;
typedef __attribute__((ext_vector_type(16))) _Float16 v16h;
typedef __attribute__((ext_vector_type(8)))  float    v8f;

#define BATCH 8
#define CDIM 256
#define IDIM 128
#define NDIM 4096
#define BN_EPS 1e-5f

union FragU { v16h h; uint4 q[2]; };
union PackU { f16 h[2]; unsigned int u; };

// A-matrix fragment (16x32 f16, MxK). ISA 7.12.2: lane L<16 holds row M=L,
// K = {k..k+7, k+16..k+23}; lane L+16 holds K = {k+8..k+15, k+24..k+31}.
__device__ __forceinline__ v16h load_a_frag(const f16* base, int row, int stride,
                                            int kbase, int grp) {
  const uint4* p = reinterpret_cast<const uint4*>(base + (size_t)row * stride + kbase + grp * 8);
  FragU u; u.q[0] = p[0]; u.q[1] = p[2];
  return u.h;
}

// B-matrix fragment (32x16 f16, KxN). Lanes 0-15 hold K={k..k+7, k+16..k+23} for col N=lane,
// lanes 16-31 hold K={k+8..k+15, k+24..k+31} (same K order as the A fragment).
// base is K-contiguous transposed storage (row index = output column).
__device__ __forceinline__ v16h load_b_frag(const f16* base, int row, int stride,
                                            int kbase, int grp) {
  const uint4* p = reinterpret_cast<const uint4*>(base + (size_t)row * stride + kbase + grp * 8);
  FragU u; u.q[0] = p[0]; u.q[1] = p[2];
  return u.h;
}

// RA = reuse-A hint: set only when the previous WMMA in program order is
// identical with the same A registers (ISA 7.12 OPSEL rules).
template <bool RA>
__device__ __forceinline__ v8f wmma_f16(v16h a, v16h b, v8f c) {
  return __builtin_amdgcn_wmma_f32_16x16x32_f16(false, a, false, b, (short)0, c,
                                                RA, false);
}

// ---------------------------------------------------------------- kernel 0 --
// f32 -> f16 weight conversion: theta_w, phi_w, g_w ([I,C]), out_w ([C,I]).
__global__ void convert_weights(const float* __restrict__ tw,
                                const float* __restrict__ pw,
                                const float* __restrict__ gw,
                                const float* __restrict__ ow,
                                f16* __restrict__ w16) {
  int idx = blockIdx.x * blockDim.x + threadIdx.x;  // 0..131071
  const int WSZ = IDIM * CDIM;                      // 32768
  const float* src;
  f16* dst;
  int off;
  if (idx < WSZ)            { src = tw; dst = w16;            off = idx; }
  else if (idx < 2 * WSZ)   { src = pw; dst = w16 + WSZ;      off = idx - WSZ; }
  else if (idx < 3 * WSZ)   { src = gw; dst = w16 + 2 * WSZ;  off = idx - 2 * WSZ; }
  else                      { src = ow; dst = w16 + 3 * WSZ;  off = idx - 3 * WSZ; }
  dst[off] = (f16)src[off];
}

// ---------------------------------------------------------------- kernel 1 --
// Projections: out[i,n] = sum_c w[i,c] * x[b,c,n] + bias[i].
// grid (64 nblocks, 8 batch, 3 proj), 256 threads = 8 waves.
// WG tile: 128 (all I) x 64 n.  theta/phi stored transposed [N,I]; g as [I,N].
__global__ __launch_bounds__(256)
void proj_kernel(const float* __restrict__ x_this, const float* __restrict__ x_other,
                 const f16* __restrict__ w16,
                 const float* __restrict__ theta_b, const float* __restrict__ phi_b,
                 const float* __restrict__ g_b,
                 f16* __restrict__ thetaT, f16* __restrict__ phiT,
                 f16* __restrict__ g16) {
  __shared__ __align__(16) f16 ldsX[64][32];  // [n][c] tile, transposed at stage
  const int tid = threadIdx.x;
  const int wave = tid >> 5, lane = tid & 31, grp = lane >> 4, col = lane & 15;
  const int n0 = blockIdx.x * 64;
  const int b = blockIdx.y;
  const int p = blockIdx.z;

  const float* src = ((p == 0) ? x_this : x_other) + (size_t)b * CDIM * NDIM;
  const f16* w = w16 + p * (IDIM * CDIM);
  const float* bias = (p == 0) ? theta_b : (p == 1 ? phi_b : g_b);
  const int i0 = wave * 16;

  v8f acc[4] = {};
  for (int c0 = 0; c0 < CDIM; c0 += 32) {
    __syncthreads();
    // stage x tile transposed + f32->f16, packed ds_store_b32
#pragma unroll
    for (int it = 0; it < 4; ++it) {
      int idx = tid + it * 256;  // 0..1023 over (64 n) x (16 c-pairs)
      int nn = idx & 63;
      int cp = idx >> 6;         // 0..15
      PackU pk;
      pk.h[0] = (f16)src[(size_t)(c0 + 2 * cp) * NDIM + n0 + nn];
      pk.h[1] = (f16)src[(size_t)(c0 + 2 * cp + 1) * NDIM + n0 + nn];
      reinterpret_cast<unsigned int*>(&ldsX[nn][0])[cp] = pk.u;
    }
    __syncthreads();
    v16h a = load_a_frag(w, i0 + col, CDIM, c0, grp);
    v16h bf[4];
#pragma unroll
    for (int nt = 0; nt < 4; ++nt)
      bf[nt] = load_b_frag(&ldsX[0][0], nt * 16 + col, 32, 0, grp);
    acc[0] = wmma_f16<false>(a, bf[0], acc[0]);
#pragma unroll
    for (int nt = 1; nt < 4; ++nt)
      acc[nt] = wmma_f16<true>(a, bf[nt], acc[nt]);
  }

  // branchless epilogue: uniform stride select, +bias, f16 store
  f16* dst;
  size_t sN, sI;
  if (p == 0)      { dst = thetaT + (size_t)b * NDIM * IDIM; sN = IDIM; sI = 1; }
  else if (p == 1) { dst = phiT + (size_t)b * NDIM * IDIM;   sN = IDIM; sI = 1; }
  else             { dst = g16 + (size_t)b * IDIM * NDIM;    sN = 1;    sI = NDIM; }
#pragma unroll
  for (int r = 0; r < 8; ++r) {
    int i = i0 + r + grp * 8;
    float bv = bias[i];
#pragma unroll
    for (int nt = 0; nt < 4; ++nt) {
      int n = n0 + nt * 16 + col;
      dst[(size_t)n * sN + (size_t)i * sI] = (f16)(acc[nt][r] + bv);
    }
  }
}

// ---------------------------------------------------------------- kernel 2 --
// Flash attention: per wave 16 query rows; keys streamed in chunks of 64.
// S = Q phi^T (16 WMMAs), online softmax, Yt += P g^T (16 WMMAs) per chunk.
// grid (32 qblocks, 8 batch), 256 threads.
__global__ __launch_bounds__(256)
void attn_kernel(const f16* __restrict__ thetaT, const f16* __restrict__ phiT,
                 const f16* __restrict__ g16, f16* __restrict__ yT) {
  __shared__ __align__(16) f16 ldsPhi[64][128];  // [m][i] 16KB
  __shared__ __align__(16) f16 ldsG[128][64];    // [i][m] 16KB
  __shared__ __align__(16) f16 ldsP[8][16][64];  // per-wave P relayout 16KB
  const int tid = threadIdx.x;
  const int wave = tid >> 5, lane = tid & 31, grp = lane >> 4, col = lane & 15;
  const int b = blockIdx.y;
  const int nq = blockIdx.x * 128 + wave * 16;

  const f16* Q = thetaT + (size_t)b * NDIM * IDIM;
  const f16* PH = phiT + (size_t)b * NDIM * IDIM;
  const f16* G = g16 + (size_t)b * IDIM * NDIM;

  // preload Q fragments (16 rows x 128 i) as 4 A-fragments
  v16h qa[4];
#pragma unroll
  for (int ks = 0; ks < 4; ++ks)
    qa[ks] = load_a_frag(Q, nq + col, IDIM, ks * 32, grp);

  v8f yt[8] = {};
  float mrun[8], lrun[8];
#pragma unroll
  for (int r = 0; r < 8; ++r) { mrun[r] = -INFINITY; lrun[r] = 0.f; }

  for (int m0 = 0; m0 < NDIM; m0 += 64) {
    __syncthreads();
    {  // stage phi rows m0..m0+63 (contiguous 1024 uint4)
      const uint4* s = reinterpret_cast<const uint4*>(PH + (size_t)m0 * IDIM);
      uint4* d = reinterpret_cast<uint4*>(&ldsPhi[0][0]);
#pragma unroll
      for (int it = 0; it < 4; ++it) d[tid + it * 256] = s[tid + it * 256];
    }
    {  // stage g columns m0..m0+63 for all 128 i rows
      uint4* d = reinterpret_cast<uint4*>(&ldsG[0][0]);
#pragma unroll
      for (int it = 0; it < 4; ++it) {
        int idx = tid + it * 256;  // 0..1023
        int ii = idx >> 3, seg = idx & 7;
        const uint4* s = reinterpret_cast<const uint4*>(G + (size_t)ii * NDIM + m0);
        d[idx] = s[seg];
      }
    }
    // prefetch next chunk into cache while this chunk computes
    if (m0 + 64 < NDIM) {
      __builtin_prefetch((const char*)(PH + (size_t)(m0 + 64) * IDIM) + tid * 64, 0, 1);
      __builtin_prefetch((const char*)(G + (size_t)(tid >> 1) * NDIM + m0 + 64) +
                             (tid & 1) * 128, 0, 1);
    }
    __syncthreads();

    // S tile: 16 n x 64 m (4 m-subtiles), B-frags hoisted for WMMA chains
    v8f sa[4] = {};
#pragma unroll
    for (int ks = 0; ks < 4; ++ks) {
      v16h bp[4];
#pragma unroll
      for (int mt = 0; mt < 4; ++mt)
        bp[mt] = load_b_frag(&ldsPhi[0][0], mt * 16 + col, 128, ks * 32, grp);
      sa[0] = wmma_f16<false>(qa[ks], bp[0], sa[0]);
#pragma unroll
      for (int mt = 1; mt < 4; ++mt)
        sa[mt] = wmma_f16<true>(qa[ks], bp[mt], sa[mt]);
    }

    // online softmax (row = (r, lane-group), columns across 16 lanes)
#pragma unroll
    for (int r = 0; r < 8; ++r) {
      float cm = fmaxf(fmaxf(sa[0][r], sa[1][r]), fmaxf(sa[2][r], sa[3][r]));
#pragma unroll
      for (int off = 1; off < 16; off <<= 1)
        cm = fmaxf(cm, __shfl_xor(cm, off, 16));
      float mnew = fmaxf(mrun[r], cm);
      float alpha = __expf(mrun[r] - mnew);
      float e0 = __expf(sa[0][r] - mnew);
      float e1 = __expf(sa[1][r] - mnew);
      float e2 = __expf(sa[2][r] - mnew);
      float e3 = __expf(sa[3][r] - mnew);
      float rs = (e0 + e1) + (e2 + e3);
#pragma unroll
      for (int off = 1; off < 16; off <<= 1)
        rs += __shfl_xor(rs, off, 16);
      lrun[r] = lrun[r] * alpha + rs;
      mrun[r] = mnew;
#pragma unroll
      for (int it = 0; it < 8; ++it) yt[it][r] *= alpha;
      // C-layout -> LDS so P can be re-read in A-fragment layout
      int pr = r + grp * 8;
      ldsP[wave][pr][col] = (f16)e0;
      ldsP[wave][pr][16 + col] = (f16)e1;
      ldsP[wave][pr][32 + col] = (f16)e2;
      ldsP[wave][pr][48 + col] = (f16)e3;
    }

    // Yt[n,i] += P[n,m] * g[i,m]; two K=32 halves, A reused across 8 i-tiles
    v16h pa0 = load_a_frag(&ldsP[wave][0][0], col, 64, 0, grp);
    v16h pa1 = load_a_frag(&ldsP[wave][0][0], col, 64, 32, grp);
    {
      v16h bg[8];
#pragma unroll
      for (int it = 0; it < 8; ++it)
        bg[it] = load_b_frag(&ldsG[0][0], it * 16 + col, 64, 0, grp);
      yt[0] = wmma_f16<false>(pa0, bg[0], yt[0]);
#pragma unroll
      for (int it = 1; it < 8; ++it) yt[it] = wmma_f16<true>(pa0, bg[it], yt[it]);
#pragma unroll
      for (int it = 0; it < 8; ++it)
        bg[it] = load_b_frag(&ldsG[0][0], it * 16 + col, 64, 32, grp);
      yt[0] = wmma_f16<false>(pa1, bg[0], yt[0]);
#pragma unroll
      for (int it = 1; it < 8; ++it) yt[it] = wmma_f16<true>(pa1, bg[it], yt[it]);
    }
  }

  // finalize: Yt /= l, store as [N, I] f16
  f16* Y = yT + (size_t)b * NDIM * IDIM;
#pragma unroll
  for (int r = 0; r < 8; ++r) {
    float inv = 1.f / lrun[r];
    int n = nq + r + grp * 8;
#pragma unroll
    for (int it = 0; it < 8; ++it)
      Y[(size_t)n * IDIM + it * 16 + col] = (f16)(yt[it][r] * inv);
  }
}

// ---------------------------------------------------------------- kernel 3 --
// z[c,n] = sum_i out_w[c,i] * y[i,n]; then bias, BN (running stats), residual.
// grid (64 nblocks, 8 batch, 2 cblocks), 256 threads; WG tile 128c x 64n.
__global__ __launch_bounds__(256)
void out_kernel(const f16* __restrict__ w16out, const f16* __restrict__ yT,
                const float* __restrict__ out_b, const float* __restrict__ gamma,
                const float* __restrict__ beta, const float* __restrict__ mean,
                const float* __restrict__ var, const float* __restrict__ x_this,
                float* __restrict__ out) {
  __shared__ __align__(16) f16 ldsY[64][128];  // 16KB
  const int tid = threadIdx.x;
  const int wave = tid >> 5, lane = tid & 31, grp = lane >> 4, col = lane & 15;
  const int n0 = blockIdx.x * 64;
  const int b = blockIdx.y;
  const int c0 = blockIdx.z * 128;

  {  // stage Yt rows n0..n0+63 (contiguous 1024 uint4)
    const uint4* s = reinterpret_cast<const uint4*>(yT + ((size_t)b * NDIM + n0) * IDIM);
    uint4* d = reinterpret_cast<uint4*>(&ldsY[0][0]);
#pragma unroll
    for (int it = 0; it < 4; ++it) d[tid + it * 256] = s[tid + it * 256];
  }
  __syncthreads();

  const int crow = c0 + wave * 16;
  v8f acc[4] = {};
#pragma unroll
  for (int ks = 0; ks < 4; ++ks) {
    v16h a = load_a_frag(w16out, crow + col, IDIM, ks * 32, grp);
    v16h bf[4];
#pragma unroll
    for (int nt = 0; nt < 4; ++nt)
      bf[nt] = load_b_frag(&ldsY[0][0], nt * 16 + col, 128, ks * 32, grp);
    acc[0] = wmma_f16<false>(a, bf[0], acc[0]);
#pragma unroll
    for (int nt = 1; nt < 4; ++nt)
      acc[nt] = wmma_f16<true>(a, bf[nt], acc[nt]);
  }

#pragma unroll
  for (int r = 0; r < 8; ++r) {
    int c = crow + r + grp * 8;
    float inv = rsqrtf(var[c] + BN_EPS);
    float sc = inv * gamma[c];
    float sh = beta[c] - mean[c] * sc;
    float bv = out_b[c];
#pragma unroll
    for (int nt = 0; nt < 4; ++nt) {
      int n = n0 + nt * 16 + col;
      size_t idx = ((size_t)b * CDIM + c) * NDIM + n;
      out[idx] = x_this[idx] + (acc[nt][r] + bv) * sc + sh;
    }
  }
}

// --------------------------------------------------------------------------
extern "C" void kernel_launch(void* const* d_in, const int* in_sizes, int n_in,
                              void* d_out, int out_size, void* d_ws, size_t ws_size,
                              hipStream_t stream) {
  const float* x_this  = (const float*)d_in[0];
  const float* x_other = (const float*)d_in[1];
  const float* theta_w = (const float*)d_in[2];
  const float* theta_b = (const float*)d_in[3];
  const float* phi_w   = (const float*)d_in[4];
  const float* phi_b   = (const float*)d_in[5];
  const float* g_w     = (const float*)d_in[6];
  const float* g_b     = (const float*)d_in[7];
  const float* out_w   = (const float*)d_in[8];
  const float* out_b   = (const float*)d_in[9];
  const float* bn_g    = (const float*)d_in[10];
  const float* bn_b    = (const float*)d_in[11];
  const float* bn_m    = (const float*)d_in[12];
  const float* bn_v    = (const float*)d_in[13];
  float* out = (float*)d_out;

  char* ws = (char*)d_ws;
  f16* w16    = (f16*)ws;                                   // 131072 f16
  f16* thetaT = (f16*)(ws + 262144);                        // [B,N,I]
  f16* phiT   = thetaT + (size_t)BATCH * NDIM * IDIM;       // [B,N,I]
  f16* g16    = phiT + (size_t)BATCH * NDIM * IDIM;         // [B,I,N]
  f16* yT     = g16 + (size_t)BATCH * NDIM * IDIM;          // [B,N,I]

  convert_weights<<<512, 256, 0, stream>>>(theta_w, phi_w, g_w, out_w, w16);
  proj_kernel<<<dim3(64, BATCH, 3), 256, 0, stream>>>(
      x_this, x_other, w16, theta_b, phi_b, g_b, thetaT, phiT, g16);
  attn_kernel<<<dim3(32, BATCH), 256, 0, stream>>>(thetaT, phiT, g16, yT);
  out_kernel<<<dim3(64, BATCH, 2), 256, 0, stream>>>(
      w16 + 3 * IDIM * CDIM, yT, out_b, bn_g, bn_b, bn_m, bn_v, x_this, out);
}
